// MultiLoraModule_48954037240345
// MI455X (gfx1250) — matrix operand order repaired and checked by reading, not compiled, not run on hardware
//
#include <hip/hip_runtime.h>

#define N_TOK    4096
#define D_IN     2048
#define D_OUT    2048
#define RANK     16
#define MAX_LORA 32

#define BM 128
#define BN 128
#define BK 32
#define NKT (D_IN / BK)
#define PITCH 56   // 112 bytes/row: 16B-aligned b128 LDS loads, conflict-free lane spread

typedef __bf16 v16bf __attribute__((ext_vector_type(16)));
typedef __bf16 v8bf  __attribute__((ext_vector_type(8)));
typedef float  v8f   __attribute__((ext_vector_type(8)));
typedef int    v4i_b __attribute__((vector_size(16)));   // b128 payload type for async-LDS builtin

#if __has_builtin(__builtin_amdgcn_global_load_async_to_lds_b128)
#define USE_ASYNC_LDS 1
#else
#define USE_ASYNC_LDS 0
#endif

// ---- workspace layout (bytes) ----
#define XHI_OFF ((size_t)0)
#define XLO_OFF ((size_t)N_TOK * D_IN * 2)              // 16 MB
#define WHI_OFF (XLO_OFF * 2)                           // 32 MB
#define WLO_OFF (WHI_OFF + (size_t)D_OUT * D_IN * 2)    // 40 MB
#define AS_OFF  (WLO_OFF + (size_t)D_OUT * D_IN * 2)    // 48 MB

// Split fp32 into hi (truncated bf16) + lo (RNE bf16 of residual): x ~= hi + lo
__device__ __forceinline__ unsigned bf_split1(float f, unsigned &lo_out) {
    unsigned u   = __float_as_uint(f);
    unsigned hu  = u & 0xFFFF0000u;
    float    res = f - __uint_as_float(hu);
    unsigned ru  = __float_as_uint(res);
    unsigned rnd = 0x7FFFu + ((ru >> 16) & 1u);
    lo_out = (ru + rnd) >> 16;
    return hu >> 16;
}

__device__ __forceinline__ void split4(float4 v, uint2 &hw, uint2 &lw) {
    unsigned l0, l1, l2, l3;
    unsigned h0 = bf_split1(v.x, l0);
    unsigned h1 = bf_split1(v.y, l1);
    unsigned h2 = bf_split1(v.z, l2);
    unsigned h3 = bf_split1(v.w, l3);
    hw.x = h0 | (h1 << 16); hw.y = h2 | (h3 << 16);
    lw.x = l0 | (l1 << 16); lw.y = l2 | (l3 << 16);
}

// Build a 16-element bf16 fragment from two 16B LDS chunks (2x ds_load_b128)
__device__ __forceinline__ v16bf ldfrag(const unsigned short* p0, const unsigned short* p1) {
    v8bf a = *(const v8bf*)p0;
    v8bf b = *(const v8bf*)p1;
    return __builtin_shufflevector(a, b, 0,1,2,3,4,5,6,7,8,9,10,11,12,13,14,15);
}

// 16B global -> LDS copy: async (ASYNCcnt, no VGPR bounce) when available
__device__ __forceinline__ void copy16(const unsigned short* g, unsigned short* l) {
#if USE_ASYNC_LDS
    __builtin_amdgcn_global_load_async_to_lds_b128(
        (__attribute__((address_space(1))) v4i_b*)g,
        (__attribute__((address_space(3))) v4i_b*)l, 0, 0);
#else
    *(uint4*)l = *(const uint4*)g;
#endif
}

// ---------------------------------------------------------------------------
// Kernel 0: one-shot fp32 -> split bf16 planes (hi/lo), grid-stride, float4
// ---------------------------------------------------------------------------
__global__ __launch_bounds__(256) void split_kernel(
    const float4* __restrict__ src, uint2* __restrict__ hi,
    uint2* __restrict__ lo, int n4)
{
    int i = blockIdx.x * blockDim.x + threadIdx.x;
    const int stride = gridDim.x * blockDim.x;
    for (; i < n4; i += stride) {
        float4 v = src[i];
        uint2 h, l;
        split4(v, h, l);
        hi[i] = h;
        lo[i] = l;
    }
}

// ---------------------------------------------------------------------------
// Kernel 1: As[n][r] = scaling[a_n] * dot(x[n,:], lora_A[a_n, r, :])
// ---------------------------------------------------------------------------
__global__ __launch_bounds__(256) void lora_down_kernel(
    const float* __restrict__ x, const int* __restrict__ ids,
    const float* __restrict__ loraA, const float* __restrict__ scal,
    float* __restrict__ As)
{
    __shared__ __align__(16) float xs[D_IN];
    const int n   = blockIdx.x;
    const int tid = threadIdx.x;

    const float4* xr  = (const float4*)(x + (size_t)n * D_IN);
    float4*       xs4 = (float4*)xs;
    xs4[tid]       = xr[tid];
    xs4[tid + 256] = xr[tid + 256];
    __syncthreads();

    const int   a = ids[n];
    const float s = scal[a];
    const int wv = tid >> 5, ln = tid & 31;

    #pragma unroll
    for (int rr = 0; rr < 2; ++rr) {
        const int r = wv + rr * 8;                      // 8 waves x 2 = 16 ranks
        const float* Ar = loraA + ((size_t)a * RANK + r) * D_IN;
        float acc = 0.f;
        for (int k = ln * 4; k < D_IN; k += 128) {
            float4 av = *(const float4*)(Ar + k);
            float4 xv = *(const float4*)(&xs[k]);
            acc += av.x * xv.x + av.y * xv.y + av.z * xv.z + av.w * xv.w;
        }
        #pragma unroll
        for (int off = 16; off > 0; off >>= 1)
            acc += __shfl_xor(acc, off, 32);
        if (ln == 0) As[(size_t)n * RANK + r] = acc * s;
    }
}

// ---------------------------------------------------------------------------
// Kernel 2: out = x @ W^T + b + As @ lora_B^T, from pre-split bf16 planes.
// 128x128 tile / 256 threads (8 waves), wave tile 32x64 = 2x4 WMMA tiles.
// Double-buffered LDS; async global->LDS copies overlap WMMA compute.
// ---------------------------------------------------------------------------
__global__ __launch_bounds__(256) void lora_gemm_kernel(
    const unsigned short* __restrict__ xhi_g, const unsigned short* __restrict__ xlo_g,
    const unsigned short* __restrict__ whi_g, const unsigned short* __restrict__ wlo_g,
    const float* __restrict__ bias, const int* __restrict__ ids,
    const float* __restrict__ loraB, const float* __restrict__ As,
    float* __restrict__ out)
{
    __shared__ __align__(16) unsigned short xs_hi[2][BM][PITCH];
    __shared__ __align__(16) unsigned short xs_lo[2][BM][PITCH];
    __shared__ __align__(16) unsigned short ws_hi[2][BN][PITCH];
    __shared__ __align__(16) unsigned short ws_lo[2][BN][PITCH];

    const int tid     = threadIdx.x;
    const int rowBase = blockIdx.y * BM;
    const int colBase = blockIdx.x * BN;

    const int lane  = tid & 31;
    const int wv    = tid >> 5;
    const int waveM = wv & 3;    // 4 waves along M (32 rows each)
    const int waveN = wv >> 2;   // 2 waves along N (64 cols each)
    const int hf    = lane >> 4; // wave half (WMMA lane-layout)
    const int lr    = lane & 15;

    // Per-thread copy slots: 2 chunks x 4 planes = 8 async b128 per iteration
    const int cr0 = tid >> 2,         ck0 = (tid & 3) << 3;
    const int cr1 = (tid + 256) >> 2, ck1 = ((tid + 256) & 3) << 3;
    const size_t g0 = (size_t)cr0 * D_IN + ck0;
    const size_t g1 = (size_t)cr1 * D_IN + ck1;
    const size_t rowOff = (size_t)rowBase * D_IN;
    const size_t colOff = (size_t)colBase * D_IN;

    const unsigned short* pxh0 = xhi_g + rowOff + g0;
    const unsigned short* pxh1 = xhi_g + rowOff + g1;
    const unsigned short* pxl0 = xlo_g + rowOff + g0;
    const unsigned short* pxl1 = xlo_g + rowOff + g1;
    const unsigned short* pwh0 = whi_g + colOff + g0;
    const unsigned short* pwh1 = whi_g + colOff + g1;
    const unsigned short* pwl0 = wlo_g + colOff + g0;
    const unsigned short* pwl1 = wlo_g + colOff + g1;

    auto issue_copies = [&](int buf) {
        copy16(pxh0, &xs_hi[buf][cr0][ck0]);
        copy16(pxh1, &xs_hi[buf][cr1][ck1]);
        copy16(pxl0, &xs_lo[buf][cr0][ck0]);
        copy16(pxl1, &xs_lo[buf][cr1][ck1]);
        copy16(pwh0, &ws_hi[buf][cr0][ck0]);
        copy16(pwh1, &ws_hi[buf][cr1][ck1]);
        copy16(pwl0, &ws_lo[buf][cr0][ck0]);
        copy16(pwl1, &ws_lo[buf][cr1][ck1]);
        pxh0 += BK; pxh1 += BK; pxl0 += BK; pxl1 += BK;
        pwh0 += BK; pwh1 += BK; pwl0 += BK; pwl1 += BK;
    };

    v8f zacc = {};
    v8f acc[2][4];
    #pragma unroll
    for (int mt = 0; mt < 2; ++mt)
        #pragma unroll
        for (int nt = 0; nt < 4; ++nt)
            acc[mt][nt] = zacc;

    issue_copies(0);   // prologue: fill buffer 0

    for (int kt = 0; kt < NKT; ++kt) {
        const int cur = kt & 1;
        // Current buffer's copies were issued one compute-phase ago -> ~free wait
#if USE_ASYNC_LDS
        asm volatile("s_wait_asynccnt 0x0" ::: "memory");
#endif
        __syncthreads();   // all waves: cur copies done, prev-buffer reads done
        if (kt + 1 < NKT)
            issue_copies(cur ^ 1);   // streams into other buffer during WMMA

        // A fragments: lane half 0 -> K {0..7,16..23}, half 1 -> K {8..15,24..31}
        v16bf ah[2], al[2];
        #pragma unroll
        for (int mt = 0; mt < 2; ++mt) {
            int rA = waveM * 32 + mt * 16 + lr;
            int k0 = hf * 8;
            ah[mt] = ldfrag(&xs_hi[cur][rA][k0], &xs_hi[cur][rA][k0 + 16]);
            al[mt] = ldfrag(&xs_lo[cur][rA][k0], &xs_lo[cur][rA][k0 + 16]);
        }
        #pragma unroll
        for (int nt = 0; nt < 4; ++nt) {
            // B fragments: column = W row; half 0 -> K {0..7,16..23}, half 1 -> K {8..15,24..31}
            int rB = waveN * 64 + nt * 16 + lr;
            int k0 = hf * 8;
            v16bf bh = ldfrag(&ws_hi[cur][rB][k0], &ws_hi[cur][rB][k0 + 16]);
            v16bf bl = ldfrag(&ws_lo[cur][rB][k0], &ws_lo[cur][rB][k0 + 16]);
            #pragma unroll
            for (int mt = 0; mt < 2; ++mt) {
                acc[mt][nt] = __builtin_amdgcn_wmma_f32_16x16x32_bf16(
                    false, ah[mt], false, bh, (short)0, acc[mt][nt], false, false);
                acc[mt][nt] = __builtin_amdgcn_wmma_f32_16x16x32_bf16(
                    false, ah[mt], false, bl, (short)0, acc[mt][nt], false, false);
                acc[mt][nt] = __builtin_amdgcn_wmma_f32_16x16x32_bf16(
                    false, al[mt], false, bh, (short)0, acc[mt][nt], false, false);
            }
        }
    }

    // Epilogue: out[n][o] = acc + bias[o] + dot16(As[n,:], lora_B[a_n, o, :])
    // C/D layout: VGPR v -> M = v + 8*hf, N = lr
    float bb[4];
    #pragma unroll
    for (int nt = 0; nt < 4; ++nt)
        bb[nt] = bias[colBase + waveN * 64 + nt * 16 + lr];

    #pragma unroll
    for (int mt = 0; mt < 2; ++mt) {
        #pragma unroll
        for (int v = 0; v < 8; ++v) {
            const int n = rowBase + waveM * 32 + mt * 16 + hf * 8 + v;
            const int a = ids[n];
            const float4* Ap = (const float4*)(As + (size_t)n * RANK);
            float4 a0 = Ap[0], a1 = Ap[1], a2 = Ap[2], a3 = Ap[3];
            #pragma unroll
            for (int nt = 0; nt < 4; ++nt) {
                const int o = colBase + waveN * 64 + nt * 16 + lr;
                const float4* Bp = (const float4*)(loraB + ((size_t)a * D_OUT + o) * RANK);
                float4 b0 = Bp[0], b1 = Bp[1], b2 = Bp[2], b3 = Bp[3];
                float add = bb[nt];
                add += a0.x * b0.x + a0.y * b0.y + a0.z * b0.z + a0.w * b0.w;
                add += a1.x * b1.x + a1.y * b1.y + a1.z * b1.z + a1.w * b1.w;
                add += a2.x * b2.x + a2.y * b2.y + a2.z * b2.z + a2.w * b2.w;
                add += a3.x * b3.x + a3.y * b3.y + a3.z * b3.z + a3.w * b3.w;
                out[(size_t)n * D_OUT + o] = acc[mt][nt][v] + add;
            }
        }
    }
}

extern "C" void kernel_launch(void* const* d_in, const int* in_sizes, int n_in,
                              void* d_out, int out_size, void* d_ws, size_t ws_size,
                              hipStream_t stream) {
    (void)in_sizes; (void)n_in; (void)out_size; (void)ws_size;
    const float* x     = (const float*)d_in[0];
    const int*   ids   = (const int*)d_in[1];
    const float* W     = (const float*)d_in[2];
    const float* bias  = (const float*)d_in[3];
    const float* loraA = (const float*)d_in[4];
    const float* loraB = (const float*)d_in[5];
    const float* scal  = (const float*)d_in[6];
    float* out = (float*)d_out;

    char* ws = (char*)d_ws;
    unsigned short* xhi = (unsigned short*)(ws + XHI_OFF);
    unsigned short* xlo = (unsigned short*)(ws + XLO_OFF);
    unsigned short* whi = (unsigned short*)(ws + WHI_OFF);
    unsigned short* wlo = (unsigned short*)(ws + WLO_OFF);
    float*          As  = (float*)(ws + AS_OFF);

    const int n4x = N_TOK * D_IN / 4;
    const int n4w = D_OUT * D_IN / 4;
    split_kernel<<<dim3(1024), 256, 0, stream>>>(
        (const float4*)x, (uint2*)xhi, (uint2*)xlo, n4x);
    split_kernel<<<dim3(1024), 256, 0, stream>>>(
        (const float4*)W, (uint2*)whi, (uint2*)wlo, n4w);
    lora_down_kernel<<<dim3(N_TOK), 256, 0, stream>>>(x, ids, loraA, scal, As);
    lora_gemm_kernel<<<dim3(D_OUT / BN, N_TOK / BM), 256, 0, stream>>>(
        xhi, xlo, whi, wlo, bias, ids, loraB, As, out);
}
